// TransformerXLDecoderLayer_27539330302101
// MI455X (gfx1250) — matrix operand order repaired and checked by reading, not compiled, not run on hardware
//
#include <hip/hip_runtime.h>
#include <hip/hip_bf16.h>

// ---------------------------------------------------------------------------
// Transformer-XL decoder layer for gfx1250 (MI455X), bf16 WMMA everywhere,
// with async global->LDS staging of the B tile (CDNA5 ASYNCcnt path).
// QLEN=MLEN=1024, KLEN=2048, BSZ=2, D_MODEL=1024, H=16, D_HEAD=64, D_FF=4096
// ---------------------------------------------------------------------------

#define QLEN   1024
#define MLEN   1024
#define KLEN   2048
#define BSZ    2
#define DMODEL 1024
#define NHEAD  16
#define DHEAD  64
#define DFF    4096

typedef __bf16 bf16_t;
typedef __attribute__((ext_vector_type(16))) __bf16 v16bf;
typedef __attribute__((ext_vector_type(8)))  float  v8f;

// ---------------- WMMA fragment loaders (CDNA5 16-bit layouts) --------------
// A (M x K row-major), 16x32 tile: lanes 0-15 -> M=lane, K = {0..7} in v0-3,
// {16..23} in v4-7; lanes 16-31 -> same M, K = {8..15, 24..31}.
__device__ __forceinline__ v16bf load_a_frag(const bf16_t* base, int ld, int lane) {
    int m = lane & 15, half = lane >> 4;
    const bf16_t* p = base + (size_t)m * ld + half * 8;
    union { uint4 q[2]; v16bf v; } f;
    f.q[0] = *(const uint4*)(p);
    f.q[1] = *(const uint4*)(p + 16);
    return f.v;
}
// B fragment read from LDS-staged Bt tile (row n holds K-slice of 32 elems,
// padded row stride): lanes 0-15 -> N=lane, K={0..7,16..23}; lanes 16-31 -> K={8..15,24..31}.
#define BROW_PAD 40   // 32 data + 8 pad bf16 elems per row (80B, 16B aligned)
__device__ __forceinline__ v16bf load_b_frag_lds(const bf16_t* buf, int nt, int lane) {
    int n = lane & 15, half = lane >> 4;
    const bf16_t* p = buf + (nt * 16 + n) * BROW_PAD + half * 8;
    union { uint4 q[2]; v16bf v; } f;
    f.q[0] = *(const uint4*)(p);
    f.q[1] = *(const uint4*)(p + 16);
    return f.v;
}

// Async global->LDS copy of one 16B chunk (ISA GLOBAL_LOAD_ASYNC_TO_LDS_B128,
// GV mode, tracked by ASYNCcnt). LDS address = low 32 bits of generic pointer.
__device__ __forceinline__ void async_copy_b128(const bf16_t* gsrc, bf16_t* lds_dst) {
    uint32_t lds_addr = (uint32_t)(uintptr_t)lds_dst;
    asm volatile("global_load_async_to_lds_b128 %0, %1, off"
                 :: "v"(lds_addr), "v"(gsrc) : "memory");
}
__device__ __forceinline__ void wait_async0() {
    asm volatile("s_wait_asynccnt 0x0" ::: "memory");
}

// Stage one 64(N) x 32(K) bf16 B tile into LDS: 256 threads x 16B = 4KB.
__device__ __forceinline__ void stage_b(const bf16_t* Bt, int ldbt, int k0,
                                        bf16_t* buf, int tid) {
    int row = tid >> 2, ch = tid & 3;             // 64 rows x 4 chunks
    async_copy_b128(Bt + (size_t)row * ldbt + k0 + ch * 8,
                    buf + row * BROW_PAD + ch * 8);
}

// ---------------- Generic bf16 GEMM: C = A(MxK) * Bt(NxK)^T -----------------
// One wave -> 16x64 output strip (4 accumulators share one A fragment); the
// 64-wide B tile is staged once per block into double-buffered LDS via async
// copies, so all 8 waves share it. blockDim=(32,8); grid=(N/64, M/128).
#define GEMM_WAVES 8
__global__ void __launch_bounds__(256)
gemm_bf16(const bf16_t* __restrict__ A, int lda,
          const bf16_t* __restrict__ Bt, int ldbt,
          const float* __restrict__ bias,
          const float* __restrict__ resid, int ldr,
          float* __restrict__ Cf, int ldc,
          bf16_t* __restrict__ Cb, int ldcb,
          int M, int K, int relu)
{
    __shared__ bf16_t bsh[2][64 * BROW_PAD];      // 2 x 5120B = 10KB

    int lane = threadIdx.x;
    int wave = threadIdx.y;
    int tid  = wave * 32 + lane;
    int tm   = blockIdx.y * GEMM_WAVES + wave;
    bool active = (tm * 16 < M);                  // uniform per wave
    int tmc = active ? tm : (M / 16 - 1);         // clamp to avoid OOB A reads

    const bf16_t* Ab = A  + (size_t)tmc * 16 * lda;
    const bf16_t* Bb = Bt + (size_t)blockIdx.x * 64 * ldbt;

    stage_b(Bb, ldbt, 0, &bsh[0][0], tid);        // prologue stage

    v8f acc0 = {}, acc1 = {}, acc2 = {}, acc3 = {};
    int nsteps = K / 32;
    for (int s = 0; s < nsteps; ++s) {
        wait_async0();                            // buf[s&1] landed (this wave)
        __syncthreads();                          // visible to all waves
        if (s + 1 < nsteps)                       // overlap next stage
            stage_b(Bb, ldbt, (s + 1) * 32, &bsh[(s + 1) & 1][0], tid);

        const bf16_t* bufc = &bsh[s & 1][0];
        v16bf a  = load_a_frag(Ab + s * 32, lda, lane);
        v16bf b0 = load_b_frag_lds(bufc, 0, lane);
        v16bf b1 = load_b_frag_lds(bufc, 1, lane);
        v16bf b2 = load_b_frag_lds(bufc, 2, lane);
        v16bf b3 = load_b_frag_lds(bufc, 3, lane);
        acc0 = __builtin_amdgcn_wmma_f32_16x16x32_bf16(false, a, false, b0, (short)0, acc0, false, false);
        acc1 = __builtin_amdgcn_wmma_f32_16x16x32_bf16(false, a, false, b1, (short)0, acc1, false, false);
        acc2 = __builtin_amdgcn_wmma_f32_16x16x32_bf16(false, a, false, b2, (short)0, acc2, false, false);
        acc3 = __builtin_amdgcn_wmma_f32_16x16x32_bf16(false, a, false, b3, (short)0, acc3, false, false);
        __syncthreads();                          // done reading buf[s&1]
    }

    if (active) {
        int half = lane >> 4, nl = lane & 15;
        int m0 = tm * 16 + half * 8;
#pragma unroll
        for (int nt = 0; nt < 4; ++nt) {
            v8f acc = (nt == 0) ? acc0 : (nt == 1) ? acc1 : (nt == 2) ? acc2 : acc3;
            int n0 = blockIdx.x * 64 + nt * 16 + nl;
            float bv = bias ? bias[n0] : 0.0f;
#pragma unroll
            for (int r = 0; r < 8; ++r) {
                int row = m0 + r;
                float v = acc[r] + bv;
                if (resid) v += resid[(size_t)row * ldr + n0];
                if (relu)  v = fmaxf(v, 0.0f);
                if (Cf) Cf[(size_t)row * ldc  + n0] = v;
                if (Cb) Cb[(size_t)row * ldcb + n0] = (bf16_t)v;
            }
        }
    }
}

// ---------------- LayerNorm over D=1024 -> bf16 -----------------------------
__global__ void __launch_bounds__(256)
layernorm_bf16(const float* __restrict__ X, const float* __restrict__ g,
               const float* __restrict__ b, bf16_t* __restrict__ out)
{
    __shared__ float red[256];
    int row = blockIdx.x;
    const float* x = X + (size_t)row * DMODEL;
    float v[4]; float s = 0.f;
#pragma unroll
    for (int t = 0; t < 4; ++t) { v[t] = x[threadIdx.x + t * 256]; s += v[t]; }
    red[threadIdx.x] = s; __syncthreads();
    for (int k = 128; k > 0; k >>= 1) {
        if (threadIdx.x < k) red[threadIdx.x] += red[threadIdx.x + k];
        __syncthreads();
    }
    float mean = red[0] * (1.0f / DMODEL); __syncthreads();
    float s2 = 0.f;
#pragma unroll
    for (int t = 0; t < 4; ++t) { float d = v[t] - mean; s2 += d * d; }
    red[threadIdx.x] = s2; __syncthreads();
    for (int k = 128; k > 0; k >>= 1) {
        if (threadIdx.x < k) red[threadIdx.x] += red[threadIdx.x + k];
        __syncthreads();
    }
    float rstd = rsqrtf(red[0] * (1.0f / DMODEL) + 1e-5f);
#pragma unroll
    for (int t = 0; t < 4; ++t) {
        int c = threadIdx.x + t * 256;
        out[(size_t)row * DMODEL + c] = (bf16_t)((v[t] - mean) * rstd * g[c] + b[c]);
    }
}

// ---------------- f32 [K,N] -> bf16 transposed [N,K] ------------------------
__global__ void __launch_bounds__(256)
transpose_to_bf16(const float* __restrict__ W, bf16_t* __restrict__ Wt, int K, int N)
{
    int idx = blockIdx.x * 256 + threadIdx.x;
    if (idx < K * N) {
        int k = idx / N, n = idx % N;
        Wt[(size_t)n * K + k] = (bf16_t)W[idx];
    }
}

// ---------------- f32 -> bf16 plain copy ------------------------------------
__global__ void __launch_bounds__(256)
copy_to_bf16(const float* __restrict__ X, bf16_t* __restrict__ Y, int count)
{
    int idx = blockIdx.x * 256 + threadIdx.x;
    if (idx < count) Y[idx] = (bf16_t)X[idx];
}

// ---------------- Split qkv heads into WMMA-friendly bf16 buffers -----------
// heads: [KLEN*BSZ rows (r = kk*BSZ+b), 3072]  (q | k | v)
// Qw/Qr: [b][h][QLEN][64], Kb: [b][h][KLEN][64], Vt: [b][h][64][KLEN]
__global__ void __launch_bounds__(256)
prep_qkv(const float* __restrict__ heads,
         const float* __restrict__ rwb, const float* __restrict__ rrb,
         bf16_t* __restrict__ Qw, bf16_t* __restrict__ Qr,
         bf16_t* __restrict__ Kb, bf16_t* __restrict__ Vt)
{
    int idx = blockIdx.x * 256 + threadIdx.x;      // over (KLEN*BSZ)*DMODEL
    if (idx >= KLEN * BSZ * DMODEL) return;
    int r = idx >> 10;                             // row in heads
    int c = idx & (DMODEL - 1);
    int j = r >> 1, b = r & 1;
    int h = c >> 6, d = c & 63;
    const float* row = heads + (size_t)r * (3 * DMODEL);
    int bh = b * NHEAD + h;
    Kb[((size_t)bh * KLEN + j) * DHEAD + d]  = (bf16_t)row[DMODEL + c];
    Vt[((size_t)bh * DHEAD + d) * KLEN + j]  = (bf16_t)row[2 * DMODEL + c];
    if (j >= MLEN) {
        int i = j - MLEN;
        float qv = row[c];
        Qw[((size_t)bh * QLEN + i) * DHEAD + d] = (bf16_t)(qv + rwb[c]);
        Qr[((size_t)bh * QLEN + i) * DHEAD + d] = (bf16_t)(qv + rrb[c]);
    }
}

// ---------------- rel-shift + mask + softmax -> bf16 prob -------------------
// score[i,j] = (AC[i,j] + BD[i, j + QLEN-1-i]) / 8 for j <= i+MLEN, else -inf
__global__ void __launch_bounds__(256)
attn_softmax(const float* __restrict__ AC, const float* __restrict__ BD,
             bf16_t* __restrict__ prob)
{
    __shared__ float red[256];
    int i = blockIdx.x;
    int jmax = i + MLEN;                           // inclusive visible bound
    const float* ac = AC + (size_t)i * KLEN;
    const float* bd = BD + (size_t)i * KLEN + (QLEN - 1 - i);
    const float scale = 0.125f;                    // 1/sqrt(64)

    float mx = -3.0e38f;
    for (int j = threadIdx.x; j <= jmax; j += 256)
        mx = fmaxf(mx, (ac[j] + bd[j]) * scale);
    red[threadIdx.x] = mx; __syncthreads();
    for (int k = 128; k > 0; k >>= 1) {
        if (threadIdx.x < k) red[threadIdx.x] = fmaxf(red[threadIdx.x], red[threadIdx.x + k]);
        __syncthreads();
    }
    mx = red[0]; __syncthreads();

    float sum = 0.f;
    for (int j = threadIdx.x; j <= jmax; j += 256)
        sum += __expf((ac[j] + bd[j]) * scale - mx);
    red[threadIdx.x] = sum; __syncthreads();
    for (int k = 128; k > 0; k >>= 1) {
        if (threadIdx.x < k) red[threadIdx.x] += red[threadIdx.x + k];
        __syncthreads();
    }
    float inv = 1.0f / red[0];

    for (int j = threadIdx.x; j < KLEN; j += 256) {
        float p = 0.f;
        if (j <= jmax) p = __expf((ac[j] + bd[j]) * scale - mx) * inv;
        prob[(size_t)i * KLEN + j] = (bf16_t)p;
    }
}

// ---------------------------------------------------------------------------
extern "C" void kernel_launch(void* const* d_in, const int* in_sizes, int n_in,
                              void* d_out, int out_size, void* d_ws, size_t ws_size,
                              hipStream_t stream)
{
    (void)in_sizes; (void)n_in; (void)out_size; (void)ws_size;
    const float* input_ = (const float*)d_in[0];   // [QLEN,BSZ,D]
    const float* mems   = (const float*)d_in[1];   // [MLEN,BSZ,D]
    const float* pos    = (const float*)d_in[2];   // [KLEN,1,D]
    /* d_in[3] = mask (causal+mem, recomputed analytically) */
    const float* ln1g = (const float*)d_in[4];
    const float* ln1b = (const float*)d_in[5];
    const float* qkvw = (const float*)d_in[6];     // [D, 3072]
    const float* qkvb = (const float*)d_in[7];
    const float* rw   = (const float*)d_in[8];     // [D, 1024]
    const float* rwb  = (const float*)d_in[9];     // [H, DH]
    const float* rrb  = (const float*)d_in[10];
    const float* ow   = (const float*)d_in[11];    // [1024, D]
    const float* ln2g = (const float*)d_in[12];
    const float* ln2b = (const float*)d_in[13];
    const float* ffw1 = (const float*)d_in[14];    // [D, DFF]
    const float* ffb1 = (const float*)d_in[15];
    const float* ffw2 = (const float*)d_in[16];    // [DFF, D]
    const float* ffb2 = (const float*)d_in[17];

    // -------- workspace carve-up (256B aligned) --------
    char* base = (char*)d_ws;
    size_t off = 0;
    auto alloc = [&](size_t bytes) -> char* {
        char* p = base + off;
        off = (off + bytes + 255) & ~(size_t)255;
        return p;
    };
    bf16_t* catbf  = (bf16_t*)alloc((size_t)KLEN * BSZ * DMODEL * 2);
    bf16_t* posbf  = (bf16_t*)alloc((size_t)KLEN * DMODEL * 2);
    bf16_t* qkvT   = (bf16_t*)alloc((size_t)3 * DMODEL * DMODEL * 2);
    bf16_t* rwT    = (bf16_t*)alloc((size_t)DMODEL * DMODEL * 2);
    bf16_t* owT    = (bf16_t*)alloc((size_t)DMODEL * DMODEL * 2);
    bf16_t* ff1T   = (bf16_t*)alloc((size_t)DMODEL * DFF * 2);
    bf16_t* ff2T   = (bf16_t*)alloc((size_t)DFF * DMODEL * 2);
    float*  heads  = (float*) alloc((size_t)KLEN * BSZ * 3 * DMODEL * 4);
    bf16_t* rkbf   = (bf16_t*)alloc((size_t)KLEN * DMODEL * 2);
    bf16_t* Qw     = (bf16_t*)alloc((size_t)BSZ * NHEAD * QLEN * DHEAD * 2);
    bf16_t* Qr     = (bf16_t*)alloc((size_t)BSZ * NHEAD * QLEN * DHEAD * 2);
    bf16_t* Kb     = (bf16_t*)alloc((size_t)BSZ * NHEAD * KLEN * DHEAD * 2);
    bf16_t* Vt     = (bf16_t*)alloc((size_t)BSZ * NHEAD * DHEAD * KLEN * 2);
    float*  ACb    = (float*) alloc((size_t)QLEN * KLEN * 4);
    float*  BDb    = (float*) alloc((size_t)QLEN * KLEN * 4);
    bf16_t* probbf = (bf16_t*)alloc((size_t)QLEN * KLEN * 2);
    bf16_t* attnbf = (bf16_t*)alloc((size_t)QLEN * BSZ * DMODEL * 2);
    float*  xbuf   = (float*) alloc((size_t)QLEN * BSZ * DMODEL * 4);
    bf16_t* ylnbf  = (bf16_t*)alloc((size_t)QLEN * BSZ * DMODEL * 2);
    bf16_t* h1bf   = (bf16_t*)alloc((size_t)QLEN * BSZ * DFF * 2);

    dim3 gblk(32, GEMM_WAVES);
    auto cdiv = [](int a, int b) { return (a + b - 1) / b; };

    // -------- weight conversion (transpose to bf16 [N,K]) --------
    transpose_to_bf16<<<cdiv(DMODEL * 3 * DMODEL, 256), 256, 0, stream>>>(qkvw, qkvT, DMODEL, 3 * DMODEL);
    transpose_to_bf16<<<cdiv(DMODEL * DMODEL, 256),     256, 0, stream>>>(rw,   rwT,  DMODEL, DMODEL);
    transpose_to_bf16<<<cdiv(DMODEL * DMODEL, 256),     256, 0, stream>>>(ow,   owT,  DMODEL, DMODEL);
    transpose_to_bf16<<<cdiv(DMODEL * DFF, 256),        256, 0, stream>>>(ffw1, ff1T, DMODEL, DFF);
    transpose_to_bf16<<<cdiv(DFF * DMODEL, 256),        256, 0, stream>>>(ffw2, ff2T, DFF, DMODEL);
    copy_to_bf16<<<cdiv(KLEN * DMODEL, 256), 256, 0, stream>>>(pos, posbf, KLEN * DMODEL);

    // -------- LN1 over mems (rows 0..2047) and input (rows 2048..4095) -----
    layernorm_bf16<<<MLEN * BSZ, 256, 0, stream>>>(mems,   ln1g, ln1b, catbf);
    layernorm_bf16<<<QLEN * BSZ, 256, 0, stream>>>(input_, ln1g, ln1b, catbf + (size_t)MLEN * BSZ * DMODEL);

    // -------- qkv projection: [4096,1024] x [1024,3072] -> heads f32 -------
    gemm_bf16<<<dim3(3 * DMODEL / 64, cdiv(KLEN * BSZ, 16 * GEMM_WAVES)), gblk, 0, stream>>>(
        catbf, DMODEL, qkvT, DMODEL, qkvb, nullptr, 0,
        heads, 3 * DMODEL, nullptr, 0, KLEN * BSZ, DMODEL, 0);

    // -------- r_k = pos_emb @ r_w : [2048,1024] -> bf16 --------------------
    gemm_bf16<<<dim3(DMODEL / 64, cdiv(KLEN, 16 * GEMM_WAVES)), gblk, 0, stream>>>(
        posbf, DMODEL, rwT, DMODEL, nullptr, nullptr, 0,
        nullptr, 0, rkbf, DMODEL, KLEN, DMODEL, 0);

    // -------- split heads + add biases -------------------------------------
    prep_qkv<<<cdiv(KLEN * BSZ * DMODEL, 256), 256, 0, stream>>>(heads, rwb, rrb, Qw, Qr, Kb, Vt);

    // -------- attention, per (b,h); score tiles stay in L2 -----------------
    for (int b = 0; b < BSZ; ++b) {
        for (int h = 0; h < NHEAD; ++h) {
            int bh = b * NHEAD + h;
            const bf16_t* qw = Qw + (size_t)bh * QLEN * DHEAD;
            const bf16_t* qr = Qr + (size_t)bh * QLEN * DHEAD;
            const bf16_t* kb = Kb + (size_t)bh * KLEN * DHEAD;
            const bf16_t* vt = Vt + (size_t)bh * DHEAD * KLEN;

            // AC[i,j] = Qw_i . K_j   (M=1024, N=2048, K=64)
            gemm_bf16<<<dim3(KLEN / 64, cdiv(QLEN, 16 * GEMM_WAVES)), gblk, 0, stream>>>(
                qw, DHEAD, kb, DHEAD, nullptr, nullptr, 0,
                ACb, KLEN, nullptr, 0, QLEN, DHEAD, 0);
            // BD_raw[i,jj] = Qr_i . r_k[jj]  (r_k slice of full [2048,1024])
            gemm_bf16<<<dim3(KLEN / 64, cdiv(QLEN, 16 * GEMM_WAVES)), gblk, 0, stream>>>(
                qr, DHEAD, rkbf + h * DHEAD, DMODEL, nullptr, nullptr, 0,
                BDb, KLEN, nullptr, 0, QLEN, DHEAD, 0);
            // rel-shift + mask + softmax -> bf16 prob
            attn_softmax<<<QLEN, 256, 0, stream>>>(ACb, BDb, probbf);
            // attn = prob @ V : [1024,2048] x [2048,64]; scatter into
            // attnbf[(i*BSZ+b)*1024 + h*64 + d]  (ldcb = 2048)
            gemm_bf16<<<dim3(DHEAD / 64, cdiv(QLEN, 16 * GEMM_WAVES)), gblk, 0, stream>>>(
                probbf, KLEN, vt, KLEN, nullptr, nullptr, 0,
                nullptr, 0, attnbf + (size_t)b * DMODEL + h * DHEAD, BSZ * DMODEL,
                QLEN, KLEN, 0);
        }
    }

    // -------- x = input_ + attn @ o_w  (residual fused) --------------------
    gemm_bf16<<<dim3(DMODEL / 64, cdiv(QLEN * BSZ, 16 * GEMM_WAVES)), gblk, 0, stream>>>(
        attnbf, DMODEL, owT, DMODEL, nullptr, input_, DMODEL,
        xbuf, DMODEL, nullptr, 0, QLEN * BSZ, DMODEL, 0);

    // -------- LN2 -> bf16 --------------------------------------------------
    layernorm_bf16<<<QLEN * BSZ, 256, 0, stream>>>(xbuf, ln2g, ln2b, ylnbf);

    // -------- FF1: relu(y @ W1 + b1) -> bf16 -------------------------------
    gemm_bf16<<<dim3(DFF / 64, cdiv(QLEN * BSZ, 16 * GEMM_WAVES)), gblk, 0, stream>>>(
        ylnbf, DMODEL, ff1T, DMODEL, ffb1, nullptr, 0,
        nullptr, 0, h1bf, DFF, QLEN * BSZ, DMODEL, 1);

    // -------- FF2 + b2 + residual x -> d_out (f32) -------------------------
    gemm_bf16<<<dim3(DMODEL / 64, cdiv(QLEN * BSZ, 16 * GEMM_WAVES)), gblk, 0, stream>>>(
        h1bf, DFF, ff2T, DFF, ffb2, xbuf, DMODEL,
        (float*)d_out, DMODEL, nullptr, 0, QLEN * BSZ, DFF, 0);
}
